// RNN_42537356100303
// MI455X (gfx1250) — compile-verified, matrix operand order checked by reading, not run on hardware
//
#include <hip/hip_runtime.h>
#include <math.h>

typedef __bf16 bf16_t;
typedef __attribute__((ext_vector_type(16))) __bf16 v16bf;
typedef __attribute__((ext_vector_type(8)))  float  v8f;

#define Bsz 128
#define Tsz 1024
#define Fsz 24
#define Hsz 512
#define M1  1024
#define Csz 8
#define NMT 8    // batch tiles of 16 (128/16)
#define NNT 32   // N tiles of 16 (512/16)

// ---------------- fp32 -> bf16 convert ----------------
__global__ void cvt_kernel(const float* __restrict__ src, bf16_t* __restrict__ dst, int n) {
    int i = blockIdx.x * blockDim.x + threadIdx.x;
    if (i < n) dst[i] = (bf16_t)src[i];
}

// pad [rows][24] f32 -> [rows][32] bf16 (zeros in k=24..31)
__global__ void pad24_kernel(const float* __restrict__ src, bf16_t* __restrict__ dst, int rows) {
    int i = blockIdx.x * blockDim.x + threadIdx.x;
    if (i < rows * 32) {
        int k = i & 31, r = i >> 5;
        dst[i] = (k < 24) ? (bf16_t)src[r * 24 + k] : (bf16_t)0.0f;
    }
}

// ---------------- input projection GEMM ----------------
// XP fragment (t, mtile, ntile) = A[16x32*nKT] x W^T tile + bias, stored in
// WMMA C/D fragment order: frag*256 + lane*8 f32.
// A is bf16 [b][T][Kfull] row-major; W is bf16 [n][Kfull] row-major.
__global__ __launch_bounds__(256)
void proj_kernel(const bf16_t* __restrict__ A, const bf16_t* __restrict__ W,
                 const float* __restrict__ bias1, const float* __restrict__ bias2,
                 float* __restrict__ XP, int Kfull, int nKT)
{
    const int lane = threadIdx.x & 31;
    const int wave = threadIdx.x >> 5;
    const int frag = blockIdx.x * 8 + wave;          // t*256 + mtile*32 + ntile
    const int ntile = frag & 31;
    const int mtile = (frag >> 5) & 7;
    const int t     = frag >> 8;
    const int lm = lane & 15;
    const int hi = lane >> 4;

    const bf16_t* Aptr = A + ((size_t)(mtile * 16 + lm) * Tsz + t) * Kfull + hi * 16;
    const bf16_t* Wptr = W + (size_t)(ntile * 16 + lm) * Kfull + hi * 16;

    const float bv = bias1[ntile * 16 + lm] + bias2[ntile * 16 + lm];
    v8f acc = { bv, bv, bv, bv, bv, bv, bv, bv };

    for (int kt = 0; kt < nKT; ++kt) {
        v16bf a = *(const v16bf*)(Aptr + kt * 32);
        v16bf b = *(const v16bf*)(Wptr + kt * 32);
        acc = __builtin_amdgcn_wmma_f32_16x16x32_bf16(false, a, false, b,
                                                      (short)0, acc, false, false);
    }
    *(v8f*)(XP + (size_t)frag * 256 + lane * 8) = acc;
}

// ---------------- serial RNN scan ----------------
// grid = 8 (one WG per 16-row batch tile), block = 512 (16 waves).
// Wave w owns N-tiles {2w, 2w+1}; W_hh B-fragments live in VGPRs for the
// whole 1024-step loop. h is double-buffered in LDS (bf16, A-fragment layout).
__global__ __launch_bounds__(512)
void rnn_kernel(const float* __restrict__ XP, const bf16_t* __restrict__ Whh,
                bf16_t* __restrict__ Hout)
{
    __shared__ bf16_t hbuf[2][16][Hsz];              // 32 KB
    const int lane  = threadIdx.x & 31;
    const int wave  = threadIdx.x >> 5;              // 0..15
    const int mtile = blockIdx.x;                    // 0..7
    const int lm = lane & 15;
    const int hi = lane >> 4;

    // zero h_{-1}
    for (int i = threadIdx.x; i < 16 * Hsz; i += 512)
        (&hbuf[0][0][0])[i] = (bf16_t)0.0f;

    // preload register-resident W_hh fragments (2 N-tiles x 16 K-tiles)
    v16bf Bf[2][16];
#pragma unroll
    for (int j = 0; j < 2; ++j) {
        const int n = (2 * wave + j) * 16 + lm;
#pragma unroll
        for (int kt = 0; kt < 16; ++kt)
            Bf[j][kt] = *(const v16bf*)(Whh + (size_t)n * Hsz + kt * 32 + hi * 16);
    }
    __syncthreads();

    for (int t = 0; t < Tsz; ++t) {
        const int rb = t & 1, wb = (t + 1) & 1;

        v8f acc[2];
#pragma unroll
        for (int j = 0; j < 2; ++j) {
            const size_t frag = ((size_t)t * NMT + mtile) * NNT + (2 * wave + j);
            acc[j] = *(const v8f*)(XP + frag * 256 + lane * 8);
        }

#pragma unroll
        for (int kt = 0; kt < 16; ++kt) {
            v16bf a = *(const v16bf*)(&hbuf[rb][lm][kt * 32 + hi * 16]);
            acc[0] = __builtin_amdgcn_wmma_f32_16x16x32_bf16(false, a, false, Bf[0][kt],
                                                             (short)0, acc[0], false, false);
            acc[1] = __builtin_amdgcn_wmma_f32_16x16x32_bf16(false, a, false, Bf[1][kt],
                                                             (short)0, acc[1], false, false);
        }

#pragma unroll
        for (int j = 0; j < 2; ++j) {
            const int n = (2 * wave + j) * 16 + lm;
#pragma unroll
            for (int r = 0; r < 8; ++r) {
                const int m = r + hi * 8;
                bf16_t hv = (bf16_t)tanhf(acc[j][r]);
                hbuf[wb][m][n] = hv;
                Hout[((size_t)(mtile * 16 + m) * Tsz + t) * Hsz + n] = hv;
            }
        }
        __syncthreads();
    }
}

// ---------------- MLP head ----------------
__global__ void head1_kernel(const bf16_t* __restrict__ Hbuf, const float* __restrict__ W1,
                             const float* __restrict__ b1, float* __restrict__ Z)
{
    int idx = blockIdx.x * blockDim.x + threadIdx.x;   // m*1024 + j
    int j = idx & (M1 - 1), m = idx >> 10;
    const bf16_t* h = Hbuf + ((size_t)m * Tsz + (Tsz - 1)) * Hsz;
    const float*  w = W1 + (size_t)j * Hsz;
    float s = b1[j];
    for (int k = 0; k < Hsz; ++k) s += (float)h[k] * w[k];
    Z[idx] = s / (1.0f + expf(-s));                    // silu
}

__global__ void head2_kernel(const float* __restrict__ Z, const float* __restrict__ W2,
                             const float* __restrict__ b2, float* __restrict__ out)
{
    int idx = blockIdx.x * blockDim.x + threadIdx.x;   // m*8 + c
    if (idx < Bsz * Csz) {
        int c = idx & 7, m = idx >> 3;
        float s = b2[c];
        const float* z = Z + (size_t)m * M1;
        const float* w = W2 + (size_t)c * M1;
        for (int j = 0; j < M1; ++j) s += z[j] * w[j];
        out[idx] = s;
    }
}

extern "C" void kernel_launch(void* const* d_in, const int* in_sizes, int n_in,
                              void* d_out, int out_size, void* d_ws, size_t ws_size,
                              hipStream_t stream)
{
    const float* x     = (const float*)d_in[0];
    const float* W_ih0 = (const float*)d_in[1];
    const float* W_hh0 = (const float*)d_in[2];
    const float* b_ih0 = (const float*)d_in[3];
    const float* b_hh0 = (const float*)d_in[4];
    const float* W_ih1 = (const float*)d_in[5];
    const float* W_hh1 = (const float*)d_in[6];
    const float* b_ih1 = (const float*)d_in[7];
    const float* b_hh1 = (const float*)d_in[8];
    const float* W_ih2 = (const float*)d_in[9];
    const float* W_hh2 = (const float*)d_in[10];
    const float* b_ih2 = (const float*)d_in[11];
    const float* b_hh2 = (const float*)d_in[12];
    const float* W1    = (const float*)d_in[13];
    const float* b1    = (const float*)d_in[14];
    const float* W2    = (const float*)d_in[15];
    const float* b2    = (const float*)d_in[16];

    char* ws = (char*)d_ws;
    float*  XP    = (float*)ws;           ws += (size_t)Bsz * Tsz * Hsz * 4;   // 256 MB
    bf16_t* Hbuf  = (bf16_t*)ws;          ws += (size_t)Bsz * Tsz * Hsz * 2;   // 128 MB
    bf16_t* xpad  = (bf16_t*)ws;          ws += (size_t)Bsz * Tsz * 32 * 2;    // 8 MB
    bf16_t* wih0p = (bf16_t*)ws;          ws += (size_t)Hsz * 32 * 2;
    bf16_t* whh0b = (bf16_t*)ws;          ws += (size_t)Hsz * Hsz * 2;
    bf16_t* whh1b = (bf16_t*)ws;          ws += (size_t)Hsz * Hsz * 2;
    bf16_t* whh2b = (bf16_t*)ws;          ws += (size_t)Hsz * Hsz * 2;
    bf16_t* wih1b = (bf16_t*)ws;          ws += (size_t)Hsz * Hsz * 2;
    bf16_t* wih2b = (bf16_t*)ws;          ws += (size_t)Hsz * Hsz * 2;
    float*  Zbuf  = (float*)ws;           ws += (size_t)Bsz * M1 * 4;

    // ---- convert / pad weights & inputs to bf16 ----
    pad24_kernel<<<(Bsz * Tsz * 32 + 255) / 256, 256, 0, stream>>>(x, xpad, Bsz * Tsz);
    pad24_kernel<<<(Hsz * 32 + 255) / 256, 256, 0, stream>>>(W_ih0, wih0p, Hsz);
    const int nW = Hsz * Hsz;
    cvt_kernel<<<(nW + 255) / 256, 256, 0, stream>>>(W_hh0, whh0b, nW);
    cvt_kernel<<<(nW + 255) / 256, 256, 0, stream>>>(W_hh1, whh1b, nW);
    cvt_kernel<<<(nW + 255) / 256, 256, 0, stream>>>(W_hh2, whh2b, nW);
    cvt_kernel<<<(nW + 255) / 256, 256, 0, stream>>>(W_ih1, wih1b, nW);
    cvt_kernel<<<(nW + 255) / 256, 256, 0, stream>>>(W_ih2, wih2b, nW);

    const int projGrid = Tsz * NMT * NNT / 8;   // 8 waves per 256-thread block

    // ---- layer 0 ----
    proj_kernel<<<projGrid, 256, 0, stream>>>(xpad, wih0p, b_ih0, b_hh0, XP, 32, 1);
    rnn_kernel<<<NMT, 512, 0, stream>>>(XP, whh0b, Hbuf);
    // ---- layer 1 ----
    proj_kernel<<<projGrid, 256, 0, stream>>>(Hbuf, wih1b, b_ih1, b_hh1, XP, Hsz, 16);
    rnn_kernel<<<NMT, 512, 0, stream>>>(XP, whh1b, Hbuf);
    // ---- layer 2 ----
    proj_kernel<<<projGrid, 256, 0, stream>>>(Hbuf, wih2b, b_ih2, b_hh2, XP, Hsz, 16);
    rnn_kernel<<<NMT, 512, 0, stream>>>(XP, whh2b, Hbuf);

    // ---- head ----
    head1_kernel<<<(Bsz * M1) / 256, 256, 0, stream>>>(Hbuf, W1, b1, Zbuf);
    head2_kernel<<<(Bsz * Csz + 255) / 256, 256, 0, stream>>>(Zbuf, W2, b2, (float*)d_out);
}
